// BayesianLinear_7292854469138
// MI455X (gfx1250) — matrix operand order repaired and checked by reading, not compiled, not run on hardware
//
#include <hip/hip_runtime.h>

// Problem constants (match reference)
#define Bc 32
#define Lc 512
#define Ic 1024
#define Oc 1024

typedef __bf16 bf16;
typedef __bf16 v16bf __attribute__((ext_vector_type(16)));
typedef float  v8f   __attribute__((ext_vector_type(8)));

#define NT 64                    // O-columns (weight rows) staged per block
#define KPAD 8                   // bf16 elements of pad per LDS row (16B) -> bank-conflict-free
#define WROWSTRIDE (Ic + KPAD)   // 1032 bf16 elems = 2064 B per staged weight row
#define LDS_BYTES (2u * NT * WROWSTRIDE * 2u)  // hi + lo arrays = 264192 B (<= 320KB WGP LDS)

union BFrag { v16bf v; uint4 u[2]; };

// Build one A fragment (16x32 bf16, ISA interleaved layout) split into hi/lo bf16.
// lane(half,m): elems 0..7 = x[m, k0+half*8 .. +7], elems 8..15 = x[m, k0+16+half*8 .. +7]
__device__ __forceinline__ void make_a_frag(const float* __restrict__ xp,
                                            v16bf& ahi, v16bf& alo) {
    const float4 f0 = *(const float4*)(xp);
    const float4 f1 = *(const float4*)(xp + 4);
    const float4 f2 = *(const float4*)(xp + 16);
    const float4 f3 = *(const float4*)(xp + 20);
    const float av[16] = { f0.x, f0.y, f0.z, f0.w, f1.x, f1.y, f1.z, f1.w,
                           f2.x, f2.y, f2.z, f2.w, f3.x, f3.y, f3.z, f3.w };
#pragma unroll
    for (int e = 0; e < 16; ++e) {
        const bf16 h = (bf16)av[e];
        ahi[e] = h;
        alo[e] = (bf16)(av[e] - (float)h);
    }
}

__global__ __launch_bounds__(256, 1)
void bayes_gemm_wmma(const float* __restrict__ x,
                     const float* __restrict__ wmean,
                     const float* __restrict__ wlogvar,
                     const float* __restrict__ bias,
                     const float* __restrict__ noise,
                     float* __restrict__ out)
{
    extern __shared__ char smem[];
    bf16* whi = (bf16*)smem;                  // [NT][WROWSTRIDE]
    bf16* wlo = whi + NT * WROWSTRIDE;

    const int tid = threadIdx.x;
    const int b   = blockIdx.y;
    const int n0  = blockIdx.x * NT;

    // ---- Phase 1: sample weights for this (b, n-strip) ONCE, split hi/lo bf16 into LDS.
    // w[n,k] = mean[n,k] + noise[b,n,k] * exp(0.5*logvar[n,k]) = mean + noise * exp2(0.72134752*logvar)
    {
        const size_t nb = (size_t)b * Oc * Ic;
        for (int q = tid; q < NT * (Ic / 4); q += 256) {
            const int r  = q >> 8;            // row 0..63
            const int k4 = (q & 255) << 2;    // k 0..1020 step 4
            const size_t g = (size_t)(n0 + r) * Ic + k4;
            const float4 mu = *(const float4*)(wmean   + g);
            const float4 lv = *(const float4*)(wlogvar + g);
            const float4 ns = *(const float4*)(noise + nb + g);
            float w0 = mu.x + ns.x * exp2f(0.72134752f * lv.x);
            float w1 = mu.y + ns.y * exp2f(0.72134752f * lv.y);
            float w2 = mu.z + ns.z * exp2f(0.72134752f * lv.z);
            float w3 = mu.w + ns.w * exp2f(0.72134752f * lv.w);
            union { bf16 h[4]; uint2 u; } ph, pl;
            ph.h[0] = (bf16)w0; pl.h[0] = (bf16)(w0 - (float)ph.h[0]);
            ph.h[1] = (bf16)w1; pl.h[1] = (bf16)(w1 - (float)ph.h[1]);
            ph.h[2] = (bf16)w2; pl.h[2] = (bf16)(w2 - (float)ph.h[2]);
            ph.h[3] = (bf16)w3; pl.h[3] = (bf16)(w3 - (float)ph.h[3]);
            const int off = r * WROWSTRIDE + k4;
            *(uint2*)(whi + off) = ph.u;
            *(uint2*)(wlo + off) = pl.u;
        }
    }
    __syncthreads();

    // ---- Phase 2: GEMM. 8 waves; each wave owns TWO 16-row M slabs per iteration so the
    // B fragments pulled from LDS are reused twice (halves ds_load per wmma).
    const int wave = tid >> 5;
    const int lane = tid & 31;
    const int col  = lane & 15;
    const int half = lane >> 4;

    for (int miter = 0; miter < Lc / 256; ++miter) {   // 2 iterations of 256 rows
        const int m0 = miter * 256 + wave * 32;
        const float* xrow0 = x + ((size_t)b * Lc + (m0 + col)) * Ic;
        const float* xrow1 = xrow0 + (size_t)16 * Ic;

        v8f acc[8];   // [slab s][n-tile t] -> acc[s*4+t]
#pragma unroll
        for (int t = 0; t < 8; ++t)
#pragma unroll
            for (int e = 0; e < 8; ++e) acc[t][e] = 0.0f;

#pragma unroll 2
        for (int k0 = 0; k0 < Ic; k0 += 32) {
            v16bf ahi0, alo0, ahi1, alo1;
            make_a_frag(xrow0 + k0 + half * 8, ahi0, alo0);
            make_a_frag(xrow1 + k0 + half * 8, ahi1, alo1);

            const int kb = k0 + half * 8;  // B layout: lane half h holds K 8h..8h+7 and 16+8h..16+8h+7
#pragma unroll
            for (int t = 0; t < 4; ++t) {
                const int off = (t * 16 + col) * WROWSTRIDE + kb;
                BFrag bh, bl;
                bh.u[0] = *(const uint4*)(whi + off);
                bh.u[1] = *(const uint4*)(whi + off + 16);
                bl.u[0] = *(const uint4*)(wlo + off);
                bl.u[1] = *(const uint4*)(wlo + off + 16);
                // 3-term split product: hi*hi + lo*hi + hi*lo (lo*lo ~2^-18, dropped)
                acc[t] = __builtin_amdgcn_wmma_f32_16x16x32_bf16(false, ahi0, false, bh.v,
                                                                 (short)0, acc[t], false, false);
                acc[t] = __builtin_amdgcn_wmma_f32_16x16x32_bf16(false, alo0, false, bh.v,
                                                                 (short)0, acc[t], false, false);
                acc[t] = __builtin_amdgcn_wmma_f32_16x16x32_bf16(false, ahi0, false, bl.v,
                                                                 (short)0, acc[t], false, false);
                acc[4 + t] = __builtin_amdgcn_wmma_f32_16x16x32_bf16(false, ahi1, false, bh.v,
                                                                 (short)0, acc[4 + t], false, false);
                acc[4 + t] = __builtin_amdgcn_wmma_f32_16x16x32_bf16(false, alo1, false, bh.v,
                                                                 (short)0, acc[4 + t], false, false);
                acc[4 + t] = __builtin_amdgcn_wmma_f32_16x16x32_bf16(false, ahi1, false, bl.v,
                                                                 (short)0, acc[4 + t], false, false);
            }
        }

        // ---- Epilogue: C layout -> VGPR j holds M = j + 8*half, N = col. Add bias, store fp32.
#pragma unroll
        for (int t = 0; t < 4; ++t) {
            const int ng = n0 + t * 16 + col;
            const float bv = bias[ng];
#pragma unroll
            for (int s = 0; s < 2; ++s) {
                const int mbase = m0 + s * 16 + half * 8;
#pragma unroll
                for (int j = 0; j < 8; ++j) {
                    out[((size_t)b * Lc + (mbase + j)) * Oc + ng] = acc[s * 4 + t][j] + bv;
                }
            }
        }
    }
}

extern "C" void kernel_launch(void* const* d_in, const int* in_sizes, int n_in,
                              void* d_out, int out_size, void* d_ws, size_t ws_size,
                              hipStream_t stream) {
    (void)in_sizes; (void)n_in; (void)d_ws; (void)ws_size; (void)out_size;
    const float* x       = (const float*)d_in[0];
    const float* wmean   = (const float*)d_in[1];
    const float* wlogvar = (const float*)d_in[2];
    const float* bias    = (const float*)d_in[3];
    const float* noise   = (const float*)d_in[4];
    float* out = (float*)d_out;

    (void)hipFuncSetAttribute((const void*)bayes_gemm_wmma,
                              hipFuncAttributeMaxDynamicSharedMemorySize,
                              (int)LDS_BYTES);

    dim3 grid(Oc / NT, Bc);  // (16, 32) = 512 workgroups; one weight strip sampled once each
    bayes_gemm_wmma<<<grid, 256, LDS_BYTES, stream>>>(x, wmean, wlogvar, bias, noise, out);
}
